// _MPNNLayer_42571715837967
// MI455X (gfx1250) — compile-verified, matrix operand order checked by reading, not run on hardware
//
#include <hip/hip_runtime.h>

// ---------------------------------------------------------------------------
// MPNN layer for MI455X (gfx1250): bf16 WMMA (16x16x32) GEMMs, fused
// gather -> msgMLP -> atomic scatter-add, then fused updMLP + residual.
// ---------------------------------------------------------------------------

typedef __attribute__((ext_vector_type(16))) __bf16 v16bf;
typedef __attribute__((ext_vector_type(8)))  __bf16 v8bf;
typedef __attribute__((ext_vector_type(8)))  float  v8f;

#define LN_EPS 1e-5f
#define NNODES 100000
#define NEDGES 800000

// ---------------------------------------------------------------------------
// helpers
// ---------------------------------------------------------------------------

// Weights are pre-packed (bf16) into WMMA B-fragment order:
// fragment (nt,kt) = 32 lanes x 16 elems contiguous (32B per lane).
__device__ __forceinline__ v16bf ld_bfrag(const __bf16* __restrict__ Wf,
                                          int KT, int nt, int kt, int lane) {
  return *(const v16bf*)(Wf + ((((nt * KT + kt) << 5) + lane) << 4));
}

// Convert a contiguous run of 8 fp32 into half of an A fragment.
__device__ __forceinline__ void cvt_run8(const float* __restrict__ p,
                                         v16bf& f, int off) {
  float4 a = *(const float4*)p;
  float4 b = *(const float4*)(p + 4);
  f[off + 0] = (__bf16)a.x; f[off + 1] = (__bf16)a.y;
  f[off + 2] = (__bf16)a.z; f[off + 3] = (__bf16)a.w;
  f[off + 4] = (__bf16)b.x; f[off + 5] = (__bf16)b.y;
  f[off + 6] = (__bf16)b.z; f[off + 7] = (__bf16)b.w;
}

__device__ __forceinline__ v8f wmma_bf16(v16bf a, v16bf b, v8f c) {
  return __builtin_amdgcn_wmma_f32_16x16x32_bf16(
      /*neg_a=*/false, a, /*neg_b=*/false, b,
      /*c_mod=*/(short)0, c, /*reuse_a=*/false, /*reuse_b=*/false);
}

// ---------------------------------------------------------------------------
// prep kernels
// ---------------------------------------------------------------------------

__global__ void zero_kernel(float4* __restrict__ p, size_t n4) {
  size_t i = (size_t)blockIdx.x * blockDim.x + threadIdx.x;
  size_t s = (size_t)gridDim.x * blockDim.x;
  for (; i < n4; i += s) p[i] = make_float4(0.f, 0.f, 0.f, 0.f);
}

// Repack fp32 row-major W[K][N] into bf16 WMMA B-fragment order.
// Per ISA 16-bit B layout: lanes 0-15 -> N=n0+lane, K runs {0..7, 16..23};
// lanes 16-31 -> N=n0+lane-16, K runs {8..15, 24..31}.
__global__ void pack_weights(const float* __restrict__ W, __bf16* __restrict__ out,
                             int K, int N) {
  int total = K * N;
  int KT = K >> 5;
  for (int i = blockIdx.x * blockDim.x + threadIdx.x; i < total;
       i += gridDim.x * blockDim.x) {
    int e    = i & 15;
    int lane = (i >> 4) & 31;
    int frag = i >> 9;
    int kt = frag % KT;
    int nt = frag / KT;
    int khalf = (lane >= 16) ? 8 : 0;
    int k = (kt << 5) + khalf + (e & 7) + ((e >= 8) ? 16 : 0);
    int n = (nt << 4) + (lane & 15);
    out[i] = (__bf16)W[k * N + n];
  }
}

// ---------------------------------------------------------------------------
// edge kernel: per wave, a 16-edge tile.
//   in  : x[N][64], edge_index[2][E], edge_attr[E][32]
//   GEMM1: (16x160) @ (160x128) -> bias -> LN -> SiLU -> GEMM2: @ (128x128)
//   scatter: atomicAdd into agg[dst][128]
// ---------------------------------------------------------------------------

__device__ __forceinline__ const float* feat_ptr_edge(
    const float* __restrict__ x, const float* __restrict__ ea,
    int s, int d, int e, int k) {
  if (k < 64)  return x + ((size_t)s << 6) + k;
  if (k < 128) return x + ((size_t)d << 6) + (k - 64);
  return ea + ((size_t)e << 5) + (k - 128);
}

__global__ __launch_bounds__(256) void edge_msg_kernel(
    const float* __restrict__ x, const int* __restrict__ ei,
    const float* __restrict__ ea,
    const __bf16* __restrict__ W1f, const float* __restrict__ b1,
    const float* __restrict__ g,   const float* __restrict__ be,
    const __bf16* __restrict__ W2f, const float* __restrict__ b2,
    float* __restrict__ agg, int E, int TE) {
  __shared__ __bf16 sH[8][16 * 128];

  const int lane  = threadIdx.x & 31;
  const int w     = threadIdx.x >> 5;
  const int col   = lane & 15;
  const int khalf = (lane >> 4) << 3;  // 0 or 8
  const int gw    = blockIdx.x * 8 + w;
  const int strd  = gridDim.x * 8;

  // per-lane column parameters (N = t*16 + col), all hot in cache
  float b1r[8], gr[8], ber[8], b2r[8];
#pragma unroll
  for (int t = 0; t < 8; t++) {
    int n = t * 16 + col;
    b1r[t] = b1[n]; gr[t] = g[n]; ber[t] = be[n]; b2r[t] = b2[n];
  }
  const v8f z8 = {0.f, 0.f, 0.f, 0.f, 0.f, 0.f, 0.f, 0.f};

  for (int tile = gw; tile < TE; tile += strd) {
    const int e0  = tile << 4;
    const int myE = e0 + col;          // lanes L and L+16 mirror edge L
    const int s   = ei[myE];
    const int d   = ei[E + myE];

    // ---------------- GEMM1: 16x160 @ 160x128 ----------------
    v8f acc[8];
#pragma unroll
    for (int t = 0; t < 8; t++) acc[t] = z8;

#pragma unroll
    for (int kc = 0; kc < 5; kc++) {
      const int kA = (kc << 5) + khalf;
      v16bf a;
      cvt_run8(feat_ptr_edge(x, ea, s, d, myE, kA),      a, 0);
      cvt_run8(feat_ptr_edge(x, ea, s, d, myE, kA + 16), a, 8);
#pragma unroll
      for (int t = 0; t < 8; t++)
        acc[t] = wmma_bf16(a, ld_bfrag(W1f, 5, t, kc, lane), acc[t]);
    }

    // ---------------- bias + LayerNorm + SiLU -> LDS (bf16) ----------------
#pragma unroll
    for (int r = 0; r < 8; r++) {
      float p = 0.f, q = 0.f;
#pragma unroll
      for (int t = 0; t < 8; t++) {
        float v = acc[t][r] + b1r[t];
        acc[t][r] = v;
        p += v; q += v * v;
      }
#pragma unroll
      for (int m = 1; m < 16; m <<= 1) {
        p += __shfl_xor(p, m, 32);
        q += __shfl_xor(q, m, 32);
      }
      const float mu   = p * (1.f / 128.f);
      const float var  = q * (1.f / 128.f) - mu * mu;
      const float rstd = rsqrtf(var + LN_EPS);
#pragma unroll
      for (int t = 0; t < 8; t++) {
        float v = (acc[t][r] - mu) * rstd * gr[t] + ber[t];
        v = v / (1.f + __expf(-v));  // SiLU
        sH[w][(r + khalf) * 128 + t * 16 + col] = (__bf16)v;
      }
    }
    asm volatile("s_wait_dscnt 0" ::: "memory");

    // ---------------- GEMM2: 16x128 @ 128x128 ----------------
    v8f acc2[8];
#pragma unroll
    for (int t = 0; t < 8; t++) acc2[t] = z8;

#pragma unroll
    for (int kc = 0; kc < 4; kc++) {
      const __bf16* hp = &sH[w][col * 128 + (kc << 5) + khalf];
      v8bf lo = *(const v8bf*)hp;
      v8bf hi = *(const v8bf*)(hp + 16);
      v16bf a;
#pragma unroll
      for (int i = 0; i < 8; i++) { a[i] = lo[i]; a[8 + i] = hi[i]; }
#pragma unroll
      for (int t = 0; t < 8; t++)
        acc2[t] = wmma_bf16(a, ld_bfrag(W2f, 4, t, kc, lane), acc2[t]);
    }

    // ---------------- scatter-add into agg[dst] ----------------
#pragma unroll
    for (int r = 0; r < 8; r++) {
      const int row  = r + khalf;            // edge-in-tile
      const int drow = __shfl(d, row, 32);   // dst of that edge
#pragma unroll
      for (int t = 0; t < 8; t++) {
        float val = acc2[t][r] + b2r[t];
        atomicAdd(&agg[((size_t)drow << 7) + t * 16 + col], val);
      }
    }
  }
}

// ---------------------------------------------------------------------------
// node kernel: per wave, a 16-node tile.
//   concat(x[64], agg[128]) -> (16x192)@(192x128) -> bias -> LN -> SiLU
//   -> (16x128)@(128x64) + bias + residual -> out
// ---------------------------------------------------------------------------

__device__ __forceinline__ const float* feat_ptr_node(
    const float* __restrict__ x, const float* __restrict__ agg,
    int node, int k) {
  if (k < 64) return x + ((size_t)node << 6) + k;
  return agg + ((size_t)node << 7) + (k - 64);
}

__global__ __launch_bounds__(256) void node_upd_kernel(
    const float* __restrict__ x, const float* __restrict__ agg,
    const __bf16* __restrict__ U1f, const float* __restrict__ ub1,
    const float* __restrict__ gu,  const float* __restrict__ beu,
    const __bf16* __restrict__ U2f, const float* __restrict__ ub2,
    float* __restrict__ out, int TN) {
  __shared__ __bf16 sH[8][16 * 128];

  const int lane  = threadIdx.x & 31;
  const int w     = threadIdx.x >> 5;
  const int col   = lane & 15;
  const int khalf = (lane >> 4) << 3;
  const int gw    = blockIdx.x * 8 + w;
  const int strd  = gridDim.x * 8;

  float b1r[8], gr[8], ber[8], b2r[4];
#pragma unroll
  for (int t = 0; t < 8; t++) {
    int n = t * 16 + col;
    b1r[t] = ub1[n]; gr[t] = gu[n]; ber[t] = beu[n];
  }
#pragma unroll
  for (int t = 0; t < 4; t++) b2r[t] = ub2[t * 16 + col];
  const v8f z8 = {0.f, 0.f, 0.f, 0.f, 0.f, 0.f, 0.f, 0.f};

  for (int tile = gw; tile < TN; tile += strd) {
    const int node0  = tile << 4;
    const int mynode = node0 + col;

    // ---------------- GEMM1: 16x192 @ 192x128 ----------------
    v8f acc[8];
#pragma unroll
    for (int t = 0; t < 8; t++) acc[t] = z8;

#pragma unroll
    for (int kc = 0; kc < 6; kc++) {
      const int kA = (kc << 5) + khalf;
      v16bf a;
      cvt_run8(feat_ptr_node(x, agg, mynode, kA),      a, 0);
      cvt_run8(feat_ptr_node(x, agg, mynode, kA + 16), a, 8);
#pragma unroll
      for (int t = 0; t < 8; t++)
        acc[t] = wmma_bf16(a, ld_bfrag(U1f, 6, t, kc, lane), acc[t]);
    }

    // ---------------- bias + LN + SiLU -> LDS ----------------
#pragma unroll
    for (int r = 0; r < 8; r++) {
      float p = 0.f, q = 0.f;
#pragma unroll
      for (int t = 0; t < 8; t++) {
        float v = acc[t][r] + b1r[t];
        acc[t][r] = v;
        p += v; q += v * v;
      }
#pragma unroll
      for (int m = 1; m < 16; m <<= 1) {
        p += __shfl_xor(p, m, 32);
        q += __shfl_xor(q, m, 32);
      }
      const float mu   = p * (1.f / 128.f);
      const float var  = q * (1.f / 128.f) - mu * mu;
      const float rstd = rsqrtf(var + LN_EPS);
#pragma unroll
      for (int t = 0; t < 8; t++) {
        float v = (acc[t][r] - mu) * rstd * gr[t] + ber[t];
        v = v / (1.f + __expf(-v));
        sH[w][(r + khalf) * 128 + t * 16 + col] = (__bf16)v;
      }
    }
    asm volatile("s_wait_dscnt 0" ::: "memory");

    // ---------------- GEMM2: 16x128 @ 128x64 ----------------
    v8f acc2[4];
#pragma unroll
    for (int t = 0; t < 4; t++) acc2[t] = z8;

#pragma unroll
    for (int kc = 0; kc < 4; kc++) {
      const __bf16* hp = &sH[w][col * 128 + (kc << 5) + khalf];
      v8bf lo = *(const v8bf*)hp;
      v8bf hi = *(const v8bf*)(hp + 16);
      v16bf a;
#pragma unroll
      for (int i = 0; i < 8; i++) { a[i] = lo[i]; a[8 + i] = hi[i]; }
#pragma unroll
      for (int t = 0; t < 4; t++)
        acc2[t] = wmma_bf16(a, ld_bfrag(U2f, 4, t, kc, lane), acc2[t]);
    }

    // ---------------- residual + bias + store ----------------
#pragma unroll
    for (int t = 0; t < 4; t++) {
#pragma unroll
      for (int r = 0; r < 8; r++) {
        const int row = node0 + r + khalf;
        const int c   = t * 16 + col;
        const size_t idx = ((size_t)row << 6) + c;
        out[idx] = x[idx] + acc2[t][r] + b2r[t];
      }
    }
  }
}

// ---------------------------------------------------------------------------
// launch
// ---------------------------------------------------------------------------

extern "C" void kernel_launch(void* const* d_in, const int* in_sizes, int n_in,
                              void* d_out, int out_size, void* d_ws, size_t ws_size,
                              hipStream_t stream) {
  const float* x   = (const float*)d_in[0];
  const int*   ei  = (const int*)d_in[1];
  const float* ea  = (const float*)d_in[2];
  const float* W1  = (const float*)d_in[3];
  const float* b1  = (const float*)d_in[4];
  const float* g1  = (const float*)d_in[5];
  const float* be1 = (const float*)d_in[6];
  const float* W2  = (const float*)d_in[7];
  const float* b2  = (const float*)d_in[8];
  const float* U1  = (const float*)d_in[9];
  const float* ub1 = (const float*)d_in[10];
  const float* gu  = (const float*)d_in[11];
  const float* beu = (const float*)d_in[12];
  const float* U2  = (const float*)d_in[13];
  const float* ub2 = (const float*)d_in[14];
  float* out = (float*)d_out;

  const int N = NNODES, E = NEDGES;

  // workspace layout (all 256B aligned)
  char* ws = (char*)d_ws;
  float*  agg = (float*)ws;            size_t off = (size_t)N * 128 * 4;
  __bf16* W1f = (__bf16*)(ws + off);   off += (size_t)160 * 128 * 2;
  __bf16* W2f = (__bf16*)(ws + off);   off += (size_t)128 * 128 * 2;
  __bf16* U1f = (__bf16*)(ws + off);   off += (size_t)192 * 128 * 2;
  __bf16* U2f = (__bf16*)(ws + off);   off += (size_t)128 * 64 * 2;
  (void)ws_size; (void)in_sizes; (void)n_in; (void)out_size;

  // prep: zero aggregation buffer, pack weights to bf16 WMMA fragments
  zero_kernel<<<2048, 256, 0, stream>>>((float4*)agg, (size_t)N * 128 / 4);
  pack_weights<<<80, 256, 0, stream>>>(W1, W1f, 160, 128);
  pack_weights<<<64, 256, 0, stream>>>(W2, W2f, 128, 128);
  pack_weights<<<96, 256, 0, stream>>>(U1, U1f, 192, 128);
  pack_weights<<<32, 256, 0, stream>>>(U2, U2f, 128, 64);

  // fused message MLP + scatter-add (E/16 wave tiles)
  edge_msg_kernel<<<2048, 256, 0, stream>>>(x, ei, ea, W1f, b1, g1, be1,
                                            W2f, b2, agg, E, E / 16);
  // fused update MLP + residual (N/16 wave tiles)
  node_upd_kernel<<<800, 256, 0, stream>>>(x, agg, U1f, ub1, gu, beu,
                                           U2f, ub2, out, N / 16);
}
